// RingEncoder_79585743994953
// MI455X (gfx1250) — matrix operand order repaired and checked by reading, not compiled, not run on hardware
//
#include <hip/hip_runtime.h>

#define EPSV 1e-5f
#define NEGV -1e30f

constexpr int NB   = 8;      // batch
constexpr int NPTS = 4096;   // points per cloud
constexpr int NN   = NB * NPTS;  // 32768

typedef __attribute__((ext_vector_type(16))) _Float16 v16h;
typedef __attribute__((ext_vector_type(8)))  float    v8f;

union HVec { v16h v; _Float16 h[16]; };

// ---------------- float atomic max/min via CAS (works on global & LDS) ------
__device__ inline void atomicMaxF(float* addr, float val) {
  unsigned int* ia = (unsigned int*)addr;
  unsigned int old = __float_as_uint(*addr);
  while (__uint_as_float(old) < val) {
    unsigned int assumed = old;
    old = atomicCAS(ia, assumed, __float_as_uint(val));
    if (old == assumed) break;
  }
}
__device__ inline void atomicMinF(float* addr, float val) {
  unsigned int* ia = (unsigned int*)addr;
  unsigned int old = __float_as_uint(*addr);
  while (__uint_as_float(old) > val) {
    unsigned int assumed = old;
    old = atomicCAS(ia, assumed, __float_as_uint(val));
    if (old == assumed) break;
  }
}

// ---------------- generic fill ----------------------------------------------
__global__ void k_fill(float* __restrict__ p, float v, int n) {
  int i = blockIdx.x * blockDim.x + threadIdx.x;
  if (i < n) p[i] = v;
}

// ---------------- WMMA GEMM: Y[b, m, n] = sum_k W[m,k] * X[b,k,n] + bias[m] -
// Each wave computes one 16(M) x 32(N) output block: one A fragment, two B
// fragments, two f32 accumulators; K fully unrolled (template).
// A fragment (16x32 f16): lane<16: M=lane, K(j)=j+(j>=8?8:0); lane>=16: +8
// B fragment (32x16 f16): N=lane%16, K(j)=j+(j>=8?8:0); lane>=16: +8
// C/D (16x16 f32): VGPR i -> row i (lanes 0-15) / row i+8 (lanes 16-31)
template<bool STORE, bool REDUCE, int K, bool WROW>
__global__ __launch_bounds__(256)
void k_gemm16(float* __restrict__ Y, int ybs, int yoff,
              const float* __restrict__ X, int xbs, int xoff,
              const float* __restrict__ W, int wbs, int wms, int wks,
              const float* __restrict__ bias,
              int M,
              float* __restrict__ sums, float* __restrict__ sumsq,
              float* __restrict__ bmax, float* __restrict__ bmin) {
  const int lane = threadIdx.x & 31;
  const int lm   = lane & 15;
  const int hi   = lane >> 4;
  const unsigned gw = (blockIdx.x * blockDim.x + threadIdx.x) >> 5;
  constexpr unsigned PT = NN >> 5;            // 32-column super-tiles (1024)
  const int pt = (int)(gw % PT);
  const int mt = (int)(gw / PT);

  const int p0 = pt << 5;
  const int b  = p0 / NPTS;                    // NPTS is constexpr -> shifts
  const int n0 = p0 % NPTS;

  const float* Xb = X + b * xbs + xoff + n0 + lm;   // column for tile 0
  const float* Wb = W + b * wbs;
  const int m_a = (mt << 4) + lm;

  v8f acc0 = {}, acc1 = {};
#pragma unroll
  for (int kk = 0; kk < K; kk += 32) {
    HVec A, B0, B1;
    if (WROW) {
      // row-major weights: two contiguous 8-float runs -> 4x b128 loads
      const float* wrow = Wb + m_a * K + kk + hi * 8;
      const float4 f0 = *(const float4*)(wrow);
      const float4 f1 = *(const float4*)(wrow + 4);
      const float4 f2 = *(const float4*)(wrow + 16);
      const float4 f3 = *(const float4*)(wrow + 20);
      A.h[0]=(_Float16)f0.x;  A.h[1]=(_Float16)f0.y;
      A.h[2]=(_Float16)f0.z;  A.h[3]=(_Float16)f0.w;
      A.h[4]=(_Float16)f1.x;  A.h[5]=(_Float16)f1.y;
      A.h[6]=(_Float16)f1.z;  A.h[7]=(_Float16)f1.w;
      A.h[8]=(_Float16)f2.x;  A.h[9]=(_Float16)f2.y;
      A.h[10]=(_Float16)f2.z; A.h[11]=(_Float16)f2.w;
      A.h[12]=(_Float16)f3.x; A.h[13]=(_Float16)f3.y;
      A.h[14]=(_Float16)f3.z; A.h[15]=(_Float16)f3.w;
    } else {
#pragma unroll
      for (int j = 0; j < 16; ++j) {
        int ka = kk + j + ((j >= 8) ? 8 : 0) + (hi ? 8 : 0);
        A.h[j] = (_Float16)Wb[m_a * wms + ka * wks];
      }
    }
#pragma unroll
    for (int j = 0; j < 16; ++j) {
      int kb = kk + j + ((j >= 8) ? 8 : 0) + (hi ? 8 : 0);
      B0.h[j] = (_Float16)Xb[kb * NPTS];
      B1.h[j] = (_Float16)Xb[kb * NPTS + 16];
    }
    acc0 = __builtin_amdgcn_wmma_f32_16x16x32_f16(false, A.v, false, B0.v,
                                                  (short)0, acc0, false, false);
    acc1 = __builtin_amdgcn_wmma_f32_16x16x32_f16(false, A.v, false, B1.v,
                                                  (short)0, acc1, false, false);
  }
#pragma unroll
  for (int i = 0; i < 8; ++i) {
    int m = (mt << 4) + i + (hi << 3);
    float bsv = bias[m];
    float v0 = acc0[i] + bsv;
    float v1 = acc1[i] + bsv;
    if (STORE) {
      float* yp = Y + b * ybs + yoff + m * NPTS + n0 + lm;
      yp[0]  = v0;
      yp[16] = v1;
    }
    if (REDUCE) {
      float s  = v0 + v1;
      float s2 = v0 * v0 + v1 * v1;
      float mx = fmaxf(v0, v1), mn = fminf(v0, v1);
#pragma unroll
      for (int off = 1; off < 16; off <<= 1) {
        s  += __shfl_xor(s,  off, 16);
        s2 += __shfl_xor(s2, off, 16);
        mx = fmaxf(mx, __shfl_xor(mx, off, 16));
        mn = fminf(mn, __shfl_xor(mn, off, 16));
      }
      if (lm == 0) {
        atomicAdd(&sums[m],  s);
        atomicAdd(&sumsq[m], s2);
        atomicMaxF(&bmax[b * M + m], mx);
        atomicMinF(&bmin[b * M + m], mn);
      }
    }
  }
}

// ---------------- small-Cin pointwise conv (Cin=5) --------------------------
__global__ void k_conv_small(float* __restrict__ Y, const float* __restrict__ X,
                             const float* __restrict__ W, const float* __restrict__ bias,
                             int Co, int Ci) {
  int idx = blockIdx.x * blockDim.x + threadIdx.x;
  int tot = NB * Co * NPTS;
  if (idx >= tot) return;
  int n = idx % NPTS;
  int c = (idx / NPTS) % Co;
  int b = idx / (Co * NPTS);
  float acc = bias[c];
  for (int i = 0; i < Ci; ++i)
    acc += W[c * Ci + i] * X[(b * Ci + i) * NPTS + n];
  Y[idx] = acc;
}

// ---------------- per-channel sums over (B,N) for stored activations --------
__global__ void k_chan_stats(const float* __restrict__ Y, int C,
                             float* __restrict__ sums, float* __restrict__ sumsq) {
  int c = blockIdx.x;
  float s = 0.f, s2 = 0.f;
  for (int i = threadIdx.x; i < NN; i += blockDim.x) {
    int b = i / NPTS, n = i % NPTS;
    float v = Y[(b * C + c) * NPTS + n];
    s += v; s2 += v * v;
  }
  __shared__ float sh[256], sh2[256];
  sh[threadIdx.x] = s; sh2[threadIdx.x] = s2; __syncthreads();
  for (int off = 128; off > 0; off >>= 1) {
    if ((int)threadIdx.x < off) { sh[threadIdx.x] += sh[threadIdx.x + off];
                                  sh2[threadIdx.x] += sh2[threadIdx.x + off]; }
    __syncthreads();
  }
  if (threadIdx.x == 0) { sums[c] = sh[0]; sumsq[c] = sh2[0]; }
}

// ---------------- in-place BN(+ReLU) over (B,C,N) ---------------------------
__global__ void k_bnrelu(float* __restrict__ Y, const float* __restrict__ sums,
                         const float* __restrict__ sumsq, const float* __restrict__ g,
                         const float* __restrict__ bt, int C, int relu) {
  int idx = blockIdx.x * blockDim.x + threadIdx.x;
  int tot = NB * C * NPTS;
  if (idx >= tot) return;
  int c = (idx / NPTS) % C;
  const float cnt = (float)NN;
  float m = sums[c] / cnt;
  float v = sumsq[c] / cnt - m * m;
  float y = g[c] * (Y[idx] - m) * rsqrtf(v + EPSV) + bt[c];
  if (relu) y = fmaxf(y, 0.f);
  Y[idx] = y;
}

// ---------------- fused BN + (ReLU) + max-over-N from reduce stats ----------
__global__ void k_postmax(float* __restrict__ gmax, const float* __restrict__ sums,
                          const float* __restrict__ ssq, const float* __restrict__ bmax,
                          const float* __restrict__ bmin, const float* __restrict__ g,
                          const float* __restrict__ bt, int C, int relu) {
  int idx = blockIdx.x * blockDim.x + threadIdx.x;
  if (idx >= NB * C) return;
  int c = idx % C;
  const float cnt = (float)NN;
  float m = sums[c] / cnt;
  float v = ssq[c] / cnt - m * m;
  float a = g[c] * rsqrtf(v + EPSV);
  float dd = bt[c] - a * m;
  float base = (a >= 0.f) ? bmax[idx] : bmin[idx];
  float val = a * base + dd;
  if (relu) val = fmaxf(val, 0.f);
  gmax[idx] = val;
}

// ---------------- tiny FC layers --------------------------------------------
__global__ void k_fc(float* __restrict__ Y, const float* __restrict__ X,
                     const float* __restrict__ W, const float* __restrict__ bias,
                     int Co, int Ci) {
  int idx = blockIdx.x * blockDim.x + threadIdx.x;
  if (idx >= NB * Co) return;
  int o = idx % Co, b = idx / Co;
  float acc = bias[o];
  for (int k = 0; k < Ci; ++k) acc += X[b * Ci + k] * W[o * Ci + k];
  Y[idx] = acc;
}

__global__ void k_bnfc(float* __restrict__ Y, const float* __restrict__ g,
                       const float* __restrict__ bt, int C, int relu) {
  int c = blockIdx.x * blockDim.x + threadIdx.x;
  if (c >= C) return;
  float xs[NB]; float m = 0.f;
  for (int b = 0; b < NB; ++b) { xs[b] = Y[b * C + c]; m += xs[b]; }
  m /= (float)NB;
  float v = 0.f;
  for (int b = 0; b < NB; ++b) { float d = xs[b] - m; v += d * d; }
  v /= (float)NB;
  float a = g[c] * rsqrtf(v + EPSV);
  for (int b = 0; b < NB; ++b) {
    float y = a * (xs[b] - m) + bt[c];
    if (relu) y = fmaxf(y, 0.f);
    Y[b * C + c] = y;
  }
}

__global__ void k_fc_iden(float* __restrict__ Y, const float* __restrict__ X,
                          const float* __restrict__ W, const float* __restrict__ bias,
                          int kdim, int Ci) {
  int idx = blockIdx.x * blockDim.x + threadIdx.x;
  int KK = kdim * kdim;
  if (idx >= NB * KK) return;
  int o = idx % KK, b = idx / KK;
  float acc = bias[o];
  for (int k = 0; k < Ci; ++k) acc += X[b * Ci + k] * W[o * Ci + k];
  if ((o / kdim) == (o % kdim)) acc += 1.f;
  Y[idx] = acc;
}

// ---------------- apply 3x3 STN to xyz, pass through the other 2 channels ---
__global__ void k_transform(float* __restrict__ XP, const float* __restrict__ X,
                            const float* __restrict__ trans) {
  int idx = blockIdx.x * blockDim.x + threadIdx.x;
  if (idx >= NN) return;
  int n = idx % NPTS, b = idx / NPTS;
  const float* T = trans + b * 9;
  float p0 = X[(b * 5 + 0) * NPTS + n];
  float p1 = X[(b * 5 + 1) * NPTS + n];
  float p2 = X[(b * 5 + 2) * NPTS + n];
  for (int e = 0; e < 3; ++e)
    XP[(b * 5 + e) * NPTS + n] = p0 * T[e] + p1 * T[3 + e] + p2 * T[6 + e];
  XP[(b * 5 + 3) * NPTS + n] = X[(b * 5 + 3) * NPTS + n];
  XP[(b * 5 + 4) * NPTS + n] = X[(b * 5 + 4) * NPTS + n];
}

// ---------------- broadcast global-max features (channels 0..1023) ----------
__global__ void k_broadcast(float* __restrict__ out, const float* __restrict__ gmax) {
  int idx = blockIdx.x * blockDim.x + threadIdx.x;
  int tot = NB * 1024 * NPTS;
  if (idx >= tot) return;
  int n = idx % NPTS;
  int c = (idx / NPTS) % 1024;
  int b = idx / (1024 * NPTS);
  out[b * (1216 * NPTS) + c * NPTS + n] = gmax[b * 1024 + c];
}

// ---------------- ring pool -------------------------------------------------
__global__ void k_ring_hist(int* __restrict__ cnt, const int* __restrict__ ring) {
  int i = blockIdx.x * blockDim.x + threadIdx.x;
  if (i < NN) atomicAdd(&cnt[ring[i]], 1);
}

// per (b,c,ntile) block: conv at each point with its own ring's weights,
// accumulate per-(r,c) sum/sumsq and per-(b,r,c) max/min via LDS staging.
__global__ void k_ring_conv(const float* __restrict__ h, int hbs, int hoff,
                            const int* __restrict__ ring,
                            const float* __restrict__ Wr, const float* __restrict__ br,
                            float* __restrict__ rsum, float* __restrict__ rssq,
                            float* __restrict__ rbmax, float* __restrict__ rbmin) {
  constexpr int NT = NPTS / 256;
  int blk = blockIdx.x;
  int nt = blk % NT;
  int c  = (blk / NT) % 128;
  int b  = blk / (NT * 128);
  int n  = nt * 256 + threadIdx.x;
  int r  = ring[b * NPTS + n];
  float acc = br[r * 128 + c];
  const float* w = Wr + (r * 128 + c) * 64;
  const float* hb = h + b * hbs + hoff + n;
  for (int k = 0; k < 64; ++k)
    acc += w[k] * hb[k * NPTS];
  __shared__ float ss[16], s2[16], smx[16], smn[16];
  if (threadIdx.x < 16) { ss[threadIdx.x] = 0.f; s2[threadIdx.x] = 0.f;
                          smx[threadIdx.x] = -3.0e38f; smn[threadIdx.x] = 3.0e38f; }
  __syncthreads();
  atomicAdd(&ss[r], acc);
  atomicAdd(&s2[r], acc * acc);
  atomicMaxF(&smx[r], acc);
  atomicMinF(&smn[r], acc);
  __syncthreads();
  if (threadIdx.x < 16) {
    int rr = threadIdx.x;
    atomicAdd(&rsum[rr * 128 + c], ss[rr]);
    atomicAdd(&rssq[rr * 128 + c], s2[rr]);
    atomicMaxF(&rbmax[(b * 16 + rr) * 128 + c], smx[rr]);
    atomicMinF(&rbmin[(b * 16 + rr) * 128 + c], smn[rr]);
  }
}

__global__ void k_ring_seg(float* __restrict__ seg, const float* __restrict__ rsum,
                           const float* __restrict__ rssq, const int* __restrict__ rcnt,
                           const float* __restrict__ rbmax, const float* __restrict__ rbmin,
                           const float* __restrict__ rg, const float* __restrict__ rbt) {
  int idx = blockIdx.x * blockDim.x + threadIdx.x;
  if (idx >= NB * 16 * 128) return;
  int c = idx % 128;
  int r = (idx / 128) % 16;
  float cnt = (float)max(rcnt[r], 1);
  float m = rsum[r * 128 + c] / cnt;
  float v = rssq[r * 128 + c] / cnt - m * m;
  float a = rg[r * 128 + c] * rsqrtf(v + EPSV);
  float dd = rbt[r * 128 + c] - a * m;
  float base = (a >= 0.f) ? rbmax[idx] : rbmin[idx];
  float val = (fabsf(base) > 1e37f) ? NEGV : (a * base + dd);
  seg[idx] = val;
}

__global__ void k_ring_scatter(float* __restrict__ out, const float* __restrict__ seg,
                               const int* __restrict__ ring) {
  int idx = blockIdx.x * blockDim.x + threadIdx.x;
  int tot = NB * 128 * NPTS;
  if (idx >= tot) return;
  int n = idx % NPTS;
  int c = (idx / NPTS) % 128;
  int b = idx / (128 * NPTS);
  int r = ring[b * NPTS + n];
  out[b * (1216 * NPTS) + (1024 + c) * NPTS + n] = seg[(b * 16 + r) * 128 + c];
}

// ============================================================================
struct StnP {
  const float *c1w,*c1b,*c2w,*c2b,*c3w,*c3b,*f1w,*f1b,*f2w,*f2b,*f3w,*f3b;
  const float *g1,*bt1,*g2,*bt2,*g3,*bt3,*g4,*bt4,*g5,*bt5;
};

#define CDIV(a, b) (((a) + (b) - 1) / (b))

extern "C" void kernel_launch(void* const* d_in, const int* in_sizes, int n_in,
                              void* d_out, int out_size, void* d_ws, size_t ws_size,
                              hipStream_t stream) {
  (void)in_sizes; (void)n_in; (void)out_size; (void)ws_size;

  const float* x    = (const float*)d_in[0];
  const int*   ring = (const int*)d_in[1];
  auto IN = [&](int i) { return (const float*)d_in[i]; };
  auto getStn = [&](int q) {
    StnP s;
    s.c1w = IN(q+0);  s.c1b = IN(q+1);  s.c2w = IN(q+2);  s.c2b = IN(q+3);
    s.c3w = IN(q+4);  s.c3b = IN(q+5);  s.f1w = IN(q+6);  s.f1b = IN(q+7);
    s.f2w = IN(q+8);  s.f2b = IN(q+9);  s.f3w = IN(q+10); s.f3b = IN(q+11);
    s.g1  = IN(q+12); s.bt1 = IN(q+13); s.g2  = IN(q+14); s.bt2 = IN(q+15);
    s.g3  = IN(q+16); s.bt3 = IN(q+17); s.g4  = IN(q+18); s.bt4 = IN(q+19);
    s.g5  = IN(q+20); s.bt5 = IN(q+21);
    return s;
  };
  StnP stn  = getStn(2);
  StnP fstn = getStn(24);
  const float *c1w = IN(46), *c1b = IN(47), *g1 = IN(48), *bt1 = IN(49);
  const float *gw1 = IN(50), *gb1 = IN(51), *gg1 = IN(52), *gbt1 = IN(53);
  const float *gw2 = IN(54), *gb2 = IN(55), *gg2 = IN(56), *gbt2 = IN(57);
  const float *rw = IN(58), *rb = IN(59), *rg = IN(60), *rbt = IN(61);

  float* out = (float*)d_out;
  const int OFF_TRANS = 1216 * NN;             // 39,845,888
  const int OFF_TFEAT = OFF_TRANS + NB * 9;
  float* outTrans = out + OFF_TRANS;
  float* outTfeat = out + OFF_TFEAT;

  // workspace layout (floats)
  float* ws = (float*)d_ws;
  long off = 0;
  float* bufS1 = ws + off; off += 64L * NN;
  float* bufS2 = ws + off; off += 128L * NN;
  float* bufH0 = ws + off; off += 64L * NN;
  float* bufXP = ws + off; off += 5L * NN;
  float* gmaxB = ws + off; off += NB * 1024;
  float* f1buf = ws + off; off += NB * 512;
  float* f2buf = ws + off; off += NB * 256;
  float* sums  = ws + off; off += 1024;
  float* ssq   = ws + off; off += 1024;
  float* bmaxB = ws + off; off += NB * 1024;
  float* bminB = ws + off; off += NB * 1024;
  float* rsum  = ws + off; off += 2048;
  float* rssq  = ws + off; off += 2048;
  int*   rcnt  = (int*)(ws + off); off += 16;
  float* rbmax = ws + off; off += NB * 2048;
  float* rbmin = ws + off; off += NB * 2048;
  float* segB  = ws + off; off += NB * 2048;
  float* zeroB = ws + off; off += 64;          // zero bias stub

  auto fill = [&](float* p, float v, int n) {
    k_fill<<<CDIV(n, 256), 256, 0, stream>>>(p, v, n);
  };
  // STORE GEMM, row-major shared weights, K=64
  auto gemm_s64 = [&](float* Y, int ybs, int yoff, const float* X, int xbs, int xoff,
                      const float* W, const float* bias, int M) {
    int waves = (M / 16) * (NN / 32);
    k_gemm16<true, false, 64, true><<<waves / 8, 256, 0, stream>>>(
        Y, ybs, yoff, X, xbs, xoff, W, 0, 64, 1, bias, M,
        nullptr, nullptr, nullptr, nullptr);
  };
  // REDUCE GEMM (stats only), row-major shared weights, K=128
  auto gemm_r128 = [&](const float* X, int xbs, int xoff,
                       const float* W, const float* bias, int M) {
    fill(sums, 0.f, M); fill(ssq, 0.f, M);
    fill(bmaxB, -3.0e38f, NB * M); fill(bminB, 3.0e38f, NB * M);
    int waves = (M / 16) * (NN / 32);
    k_gemm16<false, true, 128, true><<<waves / 8, 256, 0, stream>>>(
        nullptr, 0, 0, X, xbs, xoff, W, 0, 128, 1, bias, M,
        sums, ssq, bmaxB, bminB);
  };
  auto bn_sp = [&](float* Y, int C, const float* g, const float* bt, int relu) {
    k_chan_stats<<<C, 256, 0, stream>>>(Y, C, sums, ssq);
    int tot = C * NN;
    k_bnrelu<<<CDIV(tot, 256), 256, 0, stream>>>(Y, sums, ssq, g, bt, C, relu);
  };
  auto stn_tail = [&](const StnP& s, int kdim, float* outKK) {
    k_fc<<<CDIV(NB * 512, 128), 128, 0, stream>>>(f1buf, gmaxB, s.f1w, s.f1b, 512, 1024);
    k_bnfc<<<CDIV(512, 128), 128, 0, stream>>>(f1buf, s.g4, s.bt4, 512, 1);
    k_fc<<<CDIV(NB * 256, 128), 128, 0, stream>>>(f2buf, f1buf, s.f2w, s.f2b, 256, 512);
    k_bnfc<<<CDIV(256, 128), 128, 0, stream>>>(f2buf, s.g5, s.bt5, 256, 1);
    k_fc_iden<<<CDIV(NB * kdim * kdim, 128), 128, 0, stream>>>(outKK, f2buf, s.f3w, s.f3b,
                                                               kdim, 256);
  };

  fill(zeroB, 0.f, 64);

  // ---------------- STN1 (cin=5, k=3) on x ----------------------------------
  k_conv_small<<<CDIV(64 * NN, 256), 256, 0, stream>>>(bufS1, x, stn.c1w, stn.c1b, 64, 5);
  bn_sp(bufS1, 64, stn.g1, stn.bt1, 1);
  gemm_s64(bufS2, 128 * NPTS, 0, bufS1, 64 * NPTS, 0, stn.c2w, stn.c2b, 128);
  bn_sp(bufS2, 128, stn.g2, stn.bt2, 1);
  gemm_r128(bufS2, 128 * NPTS, 0, stn.c3w, stn.c3b, 1024);
  k_postmax<<<CDIV(NB * 1024, 256), 256, 0, stream>>>(gmaxB, sums, ssq, bmaxB, bminB,
                                                      stn.g3, stn.bt3, 1024, 1);
  stn_tail(stn, 3, outTrans);

  // ---------------- apply STN, first shared conv ----------------------------
  k_transform<<<CDIV(NN, 256), 256, 0, stream>>>(bufXP, x, outTrans);
  k_conv_small<<<CDIV(64 * NN, 256), 256, 0, stream>>>(bufH0, bufXP, c1w, c1b, 64, 5);
  bn_sp(bufH0, 64, g1, bt1, 1);

  // ---------------- FSTN (cin=64, k=64) on h0 -------------------------------
  gemm_s64(bufS1, 64 * NPTS, 0, bufH0, 64 * NPTS, 0, fstn.c1w, fstn.c1b, 64);
  bn_sp(bufS1, 64, fstn.g1, fstn.bt1, 1);
  gemm_s64(bufS2, 128 * NPTS, 0, bufS1, 64 * NPTS, 0, fstn.c2w, fstn.c2b, 128);
  bn_sp(bufS2, 128, fstn.g2, fstn.bt2, 1);
  gemm_r128(bufS2, 128 * NPTS, 0, fstn.c3w, fstn.c3b, 1024);
  k_postmax<<<CDIV(NB * 1024, 256), 256, 0, stream>>>(gmaxB, sums, ssq, bmaxB, bminB,
                                                      fstn.g3, fstn.bt3, 1024, 1);
  stn_tail(fstn, 64, outTfeat);

  // ---------------- h = tfeat^T @ h0 (per batch) -> out channels 1152..1215 -
  {
    int waves = (64 / 16) * (NN / 32);
    k_gemm16<true, false, 64, false><<<waves / 8, 256, 0, stream>>>(
        out, 1216 * NPTS, 1152 * NPTS, bufH0, 64 * NPTS, 0,
        outTfeat, 64 * 64, /*wms=*/1, /*wks=*/64, zeroB, 64,
        nullptr, nullptr, nullptr, nullptr);
  }

  // ---------------- g1 = relu(bn(gw1 @ h)) ----------------------------------
  gemm_s64(bufS2, 128 * NPTS, 0, out, 1216 * NPTS, 1152 * NPTS, gw1, gb1, 128);
  bn_sp(bufS2, 128, gg1, gbt1, 1);

  // ---------------- g2 stats only; x1 = broadcast(max_n bn(g2)) -------------
  gemm_r128(bufS2, 128 * NPTS, 0, gw2, gb2, 1024);
  k_postmax<<<CDIV(NB * 1024, 256), 256, 0, stream>>>(gmaxB, sums, ssq, bmaxB, bminB,
                                                      gg2, gbt2, 1024, 0);
  k_broadcast<<<CDIV(1024 * NN, 256), 256, 0, stream>>>(out, gmaxB);

  // ---------------- ring pool -> out channels 1024..1151 --------------------
  fill(rsum, 0.f, 2048); fill(rssq, 0.f, 2048); fill((float*)rcnt, 0.f, 16);
  fill(rbmax, -3.0e38f, NB * 2048); fill(rbmin, 3.0e38f, NB * 2048);
  k_ring_hist<<<CDIV(NN, 256), 256, 0, stream>>>(rcnt, ring);
  k_ring_conv<<<NB * 128 * (NPTS / 256), 256, 0, stream>>>(
      out, 1216 * NPTS, 1152 * NPTS, ring, rw, rb, rsum, rssq, rbmax, rbmin);
  k_ring_seg<<<CDIV(NB * 16 * 128, 256), 256, 0, stream>>>(segB, rsum, rssq, rcnt,
                                                           rbmax, rbmin, rg, rbt);
  k_ring_scatter<<<CDIV(128 * NN, 256), 256, 0, stream>>>(out, segB, ring);
}
